// RNNTransformerEncoder_64037962383940
// MI455X (gfx1250) — compile-verified, matrix operand order checked by reading, not run on hardware
//
#include <hip/hip_runtime.h>
#include <hip/hip_bf16.h>
#include <math.h>

typedef __attribute__((ext_vector_type(16))) _Float16 v16h;
typedef __attribute__((ext_vector_type(8)))  float    v8f;

#define NLAYERS 6
#define NHEADS  8
#define CDIM    256
#define DMODEL  512
#define DFF     2048
#define BATCH   4
#define TLEN    2048
#define MROWS   (BATCH*TLEN)      /* 8192 */

#define USE_ASYNC_LDS 1

constexpr int BM  = 128;
constexpr int BN  = 128;
constexpr int BK  = 32;
constexpr int LDT = 40;           // LDS row stride in halfs (padded: 80B rows dodge bank conflicts)

// ---------------------------------------------------------------------------
// Async copy of 32 contiguous bytes (global f16 -> LDS), per-lane addresses.
// ISA: the instruction offset is added to both the LDS and the global address, so a
// single shared offset advances both sides.  Tracked with ASYNCcnt.
// ---------------------------------------------------------------------------
__device__ __forceinline__ void stage32_async(unsigned lds, const _Float16* g) {
#if USE_ASYNC_LDS
  unsigned long long ga = (unsigned long long)(uintptr_t)g;
  asm volatile("global_load_async_to_lds_b128 %0, %1, off\n\t"
               "global_load_async_to_lds_b128 %0, %1, off offset:16"
               :: "v"(lds), "v"(ga) : "memory");
#else
  uint4 t0 = ((const uint4*)g)[0];
  uint4 t1 = ((const uint4*)g)[1];
  _Float16* d = (_Float16*)(uintptr_t)lds;   // placeholder; sync path unused
  ((uint4*)d)[0] = t0; ((uint4*)(d + 8))[0] = t1;
#endif
}

__device__ __forceinline__ void wait_async_zero() {
#if USE_ASYNC_LDS
  asm volatile("s_wait_asynccnt 0x0" ::: "memory");
#endif
}

// ---------------------------------------------------------------------------
// LDS fragment load following the CDNA5 16-bit A/B VGPR layout:
// lanes 0-15: K=0..7 (halves 0..7), K=16..23 (halves 8..15);
// lanes 16-31: K=8..15 and K=24..31.  p = &tile[row*LDT + lgrp*8].
// ---------------------------------------------------------------------------
__device__ __forceinline__ v16h ld_frag(const _Float16* p) {
  v16h v;
  ((uint4*)&v)[0] = ((const uint4*)p)[0];        // ds_load_b128
  ((uint4*)&v)[1] = ((const uint4*)(p + 16))[0]; // ds_load_b128
  return v;
}

// ---------------------------------------------------------------------------
// Y[M,N] = act( Xh[M,K] @ Wh[N,K]^T + bias[N] ), f16 inputs, f32 accumulate.
// ACT: 0 none, 1 elu+1, 2 exact GELU.  HOUT: write f16 instead of f32.
// Double-buffered LDS fed by async-LDS DMA, 8 v_wmma per K step per wave.
// ---------------------------------------------------------------------------
template<int ACT, bool HOUT>
__global__ void __launch_bounds__(256)
gemm_wmma(const _Float16* __restrict__ Xh, const _Float16* __restrict__ Wh,
          const float* __restrict__ bias, float* __restrict__ Yf,
          _Float16* __restrict__ Yh, int M, int N, int K)
{
  __shared__ _Float16 As[2][BM * LDT];
  __shared__ _Float16 Bs[2][BN * LDT];

  const int tid  = threadIdx.x;
  const int lane = tid & 31;
  const int wid  = tid >> 5;
  const int lgrp = lane >> 4;
  const int lrow = lane & 15;
  const int rb = blockIdx.y * BM;
  const int cb = blockIdx.x * BN;
  const int wr = (wid >> 1) * 32;   // 4 waves along M
  const int wc = (wid & 1) * 64;    // 2 waves along N

  v8f acc[2][4];
  {
    v8f z = {};
    #pragma unroll
    for (int i = 0; i < 2; ++i)
      #pragma unroll
      for (int j = 0; j < 4; ++j) acc[i][j] = z;
  }

  // staging assignment: thread moves one 16-half chunk per matrix per tile
  const int sr = tid >> 1;            // tile row 0..127
  const int sc = (tid & 1) << 4;      // half-offset 0 or 16 within BK
  const _Float16* gA = Xh + (size_t)(rb + sr) * K + sc;
  const _Float16* gB = Wh + (size_t)(cb + sr) * K + sc;
  const unsigned ldsA0 = (unsigned)(uintptr_t)&As[0][sr * LDT + sc];
  const unsigned ldsA1 = (unsigned)(uintptr_t)&As[1][sr * LDT + sc];
  const unsigned ldsB0 = (unsigned)(uintptr_t)&Bs[0][sr * LDT + sc];
  const unsigned ldsB1 = (unsigned)(uintptr_t)&Bs[1][sr * LDT + sc];

  // prologue: DMA tile 0 into buffer 0
  stage32_async(ldsA0, gA);
  stage32_async(ldsB0, gB);

  for (int k0 = 0; k0 < K; k0 += BK) {
    const int cur = (k0 >> 5) & 1;
    wait_async_zero();      // this wave's DMA for tile `cur` has landed
    __syncthreads();        // everyone's landed; everyone done reading buf cur^1
    if (k0 + BK < K) {      // DMA next tile into the other buffer, overlapped
      stage32_async(cur ? ldsA0 : ldsA1, gA + k0 + BK);
      stage32_async(cur ? ldsB0 : ldsB1, gB + k0 + BK);
    }

    v16h afr[2], bfr[4];
    #pragma unroll
    for (int i = 0; i < 2; ++i)
      afr[i] = ld_frag(&As[cur][(wr + i * 16 + lrow) * LDT + lgrp * 8]);
    #pragma unroll
    for (int j = 0; j < 4; ++j)
      bfr[j] = ld_frag(&Bs[cur][(wc + j * 16 + lrow) * LDT + lgrp * 8]);

    #pragma unroll
    for (int i = 0; i < 2; ++i)
      #pragma unroll
      for (int j = 0; j < 4; ++j)
        acc[i][j] = __builtin_amdgcn_wmma_f32_16x16x32_f16(
            false, afr[i], false, bfr[j], (short)0, acc[i][j], false, false);
  }

  // epilogue: VGPR r -> lanes 0-15 M=r, lanes 16-31 M=r+8; N=lane%16
  #pragma unroll
  for (int i = 0; i < 2; ++i) {
    #pragma unroll
    for (int j = 0; j < 4; ++j) {
      const int col = cb + wc + j * 16 + lrow;
      const float bv = bias[col];
      #pragma unroll
      for (int r = 0; r < 8; ++r) {
        const int row = rb + wr + i * 16 + lgrp * 8 + r;
        float y = acc[i][j][r] + bv;
        if (ACT == 1) y = (y > 0.f) ? (y + 1.f) : __expf(y);             // elu+1
        else if (ACT == 2) y = 0.5f * y * (1.f + erff(y * 0.70710678f)); // GELU
        if (HOUT) Yh[(size_t)row * N + col] = (_Float16)y;
        else      Yf[(size_t)row * N + col] = y;
      }
    }
  }
}

// ---------------------------------------------------------------------------
// Bulk f32 -> f16 conversion (weights once per call), 8 elems/thread.
// ---------------------------------------------------------------------------
__global__ void __launch_bounds__(256)
cvt_f32_f16(const float* __restrict__ src, _Float16* __restrict__ dst)
{
  const size_t i = ((size_t)blockIdx.x * 256 + threadIdx.x) * 8;
  float4 a = ((const float4*)(src + i))[0];
  float4 b = ((const float4*)(src + i))[1];
  _Float16 h[8] = {(_Float16)a.x,(_Float16)a.y,(_Float16)a.z,(_Float16)a.w,
                   (_Float16)b.x,(_Float16)b.y,(_Float16)b.z,(_Float16)b.w};
  ((uint4*)(dst + i))[0] = ((const uint4*)h)[0];
}

// ---------------------------------------------------------------------------
// x0 = concat(transpose(x), PE) -> (B,T,DMODEL); writes f32 + f16 copies.
// ---------------------------------------------------------------------------
__global__ void __launch_bounds__(256)
prep_kernel(const float* __restrict__ xin, float* __restrict__ X0,
            _Float16* __restrict__ X0h)
{
  size_t i = (size_t)blockIdx.x * 256 + threadIdx.x;   // over B*T*D
  int d = (int)(i & (DMODEL - 1));
  size_t bt = i >> 9;
  int t = (int)(bt & (TLEN - 1));
  int b = (int)(bt >> 11);
  float v;
  if (d < CDIM) {
    v = xin[((size_t)b * CDIM + d) * TLEN + t];
  } else {
    int c = d - CDIM;
    int i2 = c >> 1;
    float freq = __expf(-0.03597788926f * (float)(2 * i2)); // exp(-ln1e4/256 * 2i)
    float ang = (float)t * freq;
    v = (c & 1) ? __cosf(ang) : __sinf(ang);
  }
  X0[i] = v;
  X0h[i] = (_Float16)v;
}

// ---------------------------------------------------------------------------
// kv[b,h,m,d] = sum_t kf[b,t,h,d]*v[b,t,h,m];  ksum[b,h,d] = sum_t kf
// ---------------------------------------------------------------------------
__global__ void __launch_bounds__(256)
kv_kernel(const float* __restrict__ Kf, const float* __restrict__ Vf,
          float* __restrict__ KV, float* __restrict__ KS)
{
  const int bh = blockIdx.x;
  const int b = bh >> 3, h = bh & 7;
  __shared__ float ks[64 * 64];
  __shared__ float vs[64 * 64];
  const int tid = threadIdx.x;
  const int d  = tid & 63;
  const int mg = tid >> 6;
  float acc[16];
  #pragma unroll
  for (int m = 0; m < 16; ++m) acc[m] = 0.f;
  float ksum = 0.f;

  const size_t base = (size_t)b * TLEN * DMODEL + h * 64;
  for (int t0 = 0; t0 < TLEN; t0 += 64) {
    #pragma unroll
    for (int rr = 0; rr < 64; rr += 4) {
      int row = rr + mg;
      size_t g = base + (size_t)(t0 + row) * DMODEL + d;
      ks[row * 64 + d] = Kf[g];
      vs[row * 64 + d] = Vf[g];
    }
    __syncthreads();
    #pragma unroll 4
    for (int tt = 0; tt < 64; ++tt) {
      float kf = ks[tt * 64 + d];
      ksum += kf;
      const float* vrow = &vs[tt * 64 + mg * 16];
      #pragma unroll
      for (int m = 0; m < 16; ++m) acc[m] += vrow[m] * kf;
    }
    __syncthreads();
  }
  #pragma unroll
  for (int m = 0; m < 16; ++m)
    KV[((size_t)bh * 64 + mg * 16 + m) * 64 + d] = acc[m];
  if (mg == 0) KS[(size_t)bh * 64 + d] = ksum;
}

// ---------------------------------------------------------------------------
// att[b,t,h,m] = (kv[b,h,m,:]·qf)/(qf·ksum+eps); writes f16 (feeds Wo GEMM)
// ---------------------------------------------------------------------------
__global__ void __launch_bounds__(64)
att_kernel(const float* __restrict__ Qf, const float* __restrict__ KV,
           const float* __restrict__ KS, _Float16* __restrict__ ATh)
{
  const int id = blockIdx.x;              // (b*T + t)*H + h
  const int h  = id & 7;
  const int bt = id >> 3;
  const int b  = bt >> 11;
  const int bh = b * NHEADS + h;
  const int m  = threadIdx.x;
  __shared__ float qs[64];
  __shared__ float ss[64];
  qs[m] = Qf[(size_t)bt * DMODEL + h * 64 + m];
  ss[m] = KS[(size_t)bh * 64 + m];
  __syncthreads();
  const float* kvrow = KV + ((size_t)bh * 64 + m) * 64;
  float zd = 0.f, acc = 0.f;
  #pragma unroll 8
  for (int d = 0; d < 64; ++d) {
    float q = qs[d];
    zd  += q * ss[d];
    acc += kvrow[d] * q;
  }
  float z = 1.f / (zd + 1e-6f);
  ATh[(size_t)bt * DMODEL + h * 64 + m] = (_Float16)(acc * z);
}

// ---------------------------------------------------------------------------
// Xio = LayerNorm(Xio + Yin)*g + be, in place; also refresh f16 copy.
// ---------------------------------------------------------------------------
__global__ void __launch_bounds__(256)
add_ln_kernel(float* __restrict__ Xio, _Float16* __restrict__ Xh,
              const float* __restrict__ Yin,
              const float* __restrict__ g, const float* __restrict__ be)
{
  __shared__ float red[256];
  const int row = blockIdx.x, tid = threadIdx.x;
  const size_t o = (size_t)row * DMODEL;
  float v0 = Xio[o + tid]       + Yin[o + tid];
  float v1 = Xio[o + tid + 256] + Yin[o + tid + 256];
  red[tid] = v0 + v1;
  __syncthreads();
  for (int s = 128; s > 0; s >>= 1) {
    if (tid < s) red[tid] += red[tid + s];
    __syncthreads();
  }
  const float mu = red[0] * (1.f / (float)DMODEL);
  __syncthreads();
  float d0 = v0 - mu, d1 = v1 - mu;
  red[tid] = d0 * d0 + d1 * d1;
  __syncthreads();
  for (int s = 128; s > 0; s >>= 1) {
    if (tid < s) red[tid] += red[tid + s];
    __syncthreads();
  }
  const float rs = rsqrtf(red[0] * (1.f / (float)DMODEL) + 1e-5f);
  float y0 = g[tid]       * d0 * rs + be[tid];
  float y1 = g[tid + 256] * d1 * rs + be[tid + 256];
  Xio[o + tid]       = y0;  Xh[o + tid]       = (_Float16)y0;
  Xio[o + tid + 256] = y1;  Xh[o + tid + 256] = (_Float16)y1;
}

// ---------------------------------------------------------------------------
// Global LN over (T,D) per batch — deterministic two-stage reduction.
// ---------------------------------------------------------------------------
__global__ void __launch_bounds__(256)
gln_partial(const float* __restrict__ X, float* __restrict__ P1, float* __restrict__ P2)
{
  const int b = blockIdx.y;
  const int tid = threadIdx.x;
  const size_t base = (size_t)b * ((size_t)TLEN * DMODEL) + (size_t)blockIdx.x * 1024;
  float s = 0.f, s2 = 0.f;
  #pragma unroll
  for (int it = 0; it < 4; ++it) {
    float v = X[base + it * 256 + tid];
    s += v; s2 += v * v;
  }
  __shared__ float r1[256], r2[256];
  r1[tid] = s; r2[tid] = s2;
  __syncthreads();
  for (int o = 128; o > 0; o >>= 1) {
    if (tid < o) { r1[tid] += r1[tid + o]; r2[tid] += r2[tid + o]; }
    __syncthreads();
  }
  if (tid == 0) { P1[b * 1024 + blockIdx.x] = r1[0]; P2[b * 1024 + blockIdx.x] = r2[0]; }
}

__global__ void __launch_bounds__(256)
gln_finish(const float* __restrict__ P1, const float* __restrict__ P2, float* __restrict__ RD)
{
  const int b = blockIdx.x, tid = threadIdx.x;
  float s = 0.f, s2 = 0.f;
  for (int i = tid; i < 1024; i += 256) { s += P1[b * 1024 + i]; s2 += P2[b * 1024 + i]; }
  __shared__ float r1[256], r2[256];
  r1[tid] = s; r2[tid] = s2;
  __syncthreads();
  for (int o = 128; o > 0; o >>= 1) {
    if (tid < o) { r1[tid] += r1[tid + o]; r2[tid] += r2[tid + o]; }
    __syncthreads();
  }
  if (tid == 0) { RD[2 * b] = r1[0]; RD[2 * b + 1] = r2[0]; }
}

__global__ void __launch_bounds__(256)
gln_apply(const float* __restrict__ X, const float* __restrict__ RD,
          const float* __restrict__ gf, const float* __restrict__ bf,
          float* __restrict__ out)
{
  size_t i = (size_t)blockIdx.x * 256 + threadIdx.x;   // over B*D*T (output order)
  const int t = (int)(i & (TLEN - 1));
  size_t bd = i >> 11;
  const int d = (int)(bd & (DMODEL - 1));
  const int b = (int)(bd >> 9);
  const float inv = 1.f / ((float)TLEN * (float)DMODEL);
  const float mu  = RD[2 * b] * inv;
  const float var = RD[2 * b + 1] * inv - mu * mu;
  const float v = X[((size_t)b * TLEN + t) * DMODEL + d];
  out[i] = gf[d] * (v - mu) * rsqrtf(var + 1e-8f) + bf[d];
}

// ---------------------------------------------------------------------------
extern "C" void kernel_launch(void* const* d_in, const int* in_sizes, int n_in,
                              void* d_out, int out_size, void* d_ws, size_t ws_size,
                              hipStream_t stream)
{
  (void)in_sizes; (void)n_in; (void)out_size; (void)ws_size;
  const float* x   = (const float*)d_in[0];
  const float* Wq  = (const float*)d_in[1];
  const float* bq  = (const float*)d_in[2];
  const float* Wk  = (const float*)d_in[3];
  const float* bk  = (const float*)d_in[4];
  const float* Wv  = (const float*)d_in[5];
  const float* bv  = (const float*)d_in[6];
  const float* Wo  = (const float*)d_in[7];
  const float* bo  = (const float*)d_in[8];
  const float* W1  = (const float*)d_in[9];
  const float* b1  = (const float*)d_in[10];
  const float* W2  = (const float*)d_in[11];
  const float* b2  = (const float*)d_in[12];
  const float* g1  = (const float*)d_in[13];
  const float* be1 = (const float*)d_in[14];
  const float* g2  = (const float*)d_in[15];
  const float* be2 = (const float*)d_in[16];
  const float* gf  = (const float*)d_in[17];
  const float* bf  = (const float*)d_in[18];
  float* out = (float*)d_out;

  float* ws = (float*)d_ws;
  const size_t MD = (size_t)MROWS * DMODEL;
  const size_t MF = (size_t)MROWS * DFF;
  const size_t WDD = (size_t)NLAYERS * DMODEL * DMODEL;   // 1,572,864
  const size_t WFD = (size_t)NLAYERS * DFF * DMODEL;      // 6,291,456

  // f32 regions
  float* X0 = ws;
  float* Qf = X0 + MD;
  float* Kf = Qf + MD;
  float* Vf = Kf + MD;
  float* Y2 = Vf + MD;
  float* KV = Y2 + MD;
  float* KS = KV + (size_t)BATCH * NHEADS * 64 * 64;
  float* P1 = KS + (size_t)BATCH * NHEADS * 64;
  float* P2 = P1 + (size_t)BATCH * 1024;
  float* RD = P2 + (size_t)BATCH * 1024;
  // f16 regions
  _Float16* hb  = (_Float16*)(RD + 8);
  _Float16* X0h = hb;
  _Float16* ATh = X0h + MD;
  _Float16* H1h = ATh + MD;
  _Float16* WQh = H1h + MF;
  _Float16* WKh = WQh + WDD;
  _Float16* WVh = WKh + WDD;
  _Float16* WOh = WVh + WDD;
  _Float16* W1h = WOh + WDD;
  _Float16* W2h = W1h + WFD;

  const dim3 blk(256);
  const dim3 gD(DMODEL / BN, MROWS / BM);         // (4, 64)
  const dim3 gF(DFF    / BN, MROWS / BM);         // (16, 64)

  // convert all weights to f16 once per call (38 MB, lives in L2 thereafter)
  cvt_f32_f16<<<(unsigned)(WDD / 2048), blk, 0, stream>>>(Wq, WQh);
  cvt_f32_f16<<<(unsigned)(WDD / 2048), blk, 0, stream>>>(Wk, WKh);
  cvt_f32_f16<<<(unsigned)(WDD / 2048), blk, 0, stream>>>(Wv, WVh);
  cvt_f32_f16<<<(unsigned)(WDD / 2048), blk, 0, stream>>>(Wo, WOh);
  cvt_f32_f16<<<(unsigned)(WFD / 2048), blk, 0, stream>>>(W1, W1h);
  cvt_f32_f16<<<(unsigned)(WFD / 2048), blk, 0, stream>>>(W2, W2h);

  prep_kernel<<<(unsigned)(MD / 256), blk, 0, stream>>>(x, X0, X0h);

  for (int i = 0; i < NLAYERS; ++i) {
    const _Float16* WQi = WQh + (size_t)i * DMODEL * DMODEL;
    const _Float16* WKi = WKh + (size_t)i * DMODEL * DMODEL;
    const _Float16* WVi = WVh + (size_t)i * DMODEL * DMODEL;
    const _Float16* WOi = WOh + (size_t)i * DMODEL * DMODEL;
    const _Float16* W1i = W1h + (size_t)i * DFF * DMODEL;
    const _Float16* W2i = W2h + (size_t)i * DMODEL * DFF;

    // projections (elu+1 feature map fused into Q/K epilogues)
    gemm_wmma<1,false><<<gD, blk, 0, stream>>>(X0h, WQi, bq + (size_t)i*DMODEL, Qf, nullptr, MROWS, DMODEL, DMODEL);
    gemm_wmma<1,false><<<gD, blk, 0, stream>>>(X0h, WKi, bk + (size_t)i*DMODEL, Kf, nullptr, MROWS, DMODEL, DMODEL);
    gemm_wmma<0,false><<<gD, blk, 0, stream>>>(X0h, WVi, bv + (size_t)i*DMODEL, Vf, nullptr, MROWS, DMODEL, DMODEL);
    // linear attention (tiny FLOPs, f32)
    kv_kernel<<<BATCH * NHEADS, blk, 0, stream>>>(Kf, Vf, KV, KS);
    att_kernel<<<MROWS * NHEADS, dim3(64), 0, stream>>>(Qf, KV, KS, ATh);
    // output projection + residual + LN1
    gemm_wmma<0,false><<<gD, blk, 0, stream>>>(ATh, WOi, bo + (size_t)i*DMODEL, Y2, nullptr, MROWS, DMODEL, DMODEL);
    add_ln_kernel<<<MROWS, blk, 0, stream>>>(X0, X0h, Y2, g1 + (size_t)i*DMODEL, be1 + (size_t)i*DMODEL);
    // FFN (W1 writes f16 directly) + residual + LN2
    gemm_wmma<2,true ><<<gF, blk, 0, stream>>>(X0h, W1i, b1 + (size_t)i*DFF, nullptr, H1h, MROWS, DFF, DMODEL);
    gemm_wmma<0,false><<<gD, blk, 0, stream>>>(H1h, W2i, b2 + (size_t)i*DMODEL, Y2, nullptr, MROWS, DMODEL, DFF);
    add_ln_kernel<<<MROWS, blk, 0, stream>>>(X0, X0h, Y2, g2 + (size_t)i*DMODEL, be2 + (size_t)i*DMODEL);
  }

  // final global LN (deterministic) + transpose to (B,D,T)
  gln_partial<<<dim3(1024, BATCH), blk, 0, stream>>>(X0, P1, P2);
  gln_finish<<<BATCH, blk, 0, stream>>>(P1, P2, RD);
  gln_apply<<<(unsigned)(MD / 256), blk, 0, stream>>>(X0, RD, gf, bf, out);
}
